// HungarianMatcher_18476949307960
// MI455X (gfx1250) — matrix operand order repaired and checked by reading, not compiled, not run on hardware
//
#include <hip/hip_runtime.h>
#include <hip/hip_bf16.h>
#include <math.h>

// Problem constants (from reference)
#define BQ   19200          // B*Q = 64*300
#define NC   92             // classes
#define NCP  96             // padded to 3 x K=32 WMMA slices
#define NT   2048           // targets
#define QT   (BQ / 16)      // 1200 query tiles
#define TTN  (NT / 16)      // 128 target tiles
#define TILES_PER_WAVE 4    // target tiles processed per wave (A/query reuse)

typedef __attribute__((ext_vector_type(16))) _Float16 v16h;
typedef __attribute__((ext_vector_type(8)))  float    v8f;

// ---------------------------------------------------------------------------
// Kernel 1: softmax over classes, emit f16 prob pre-swizzled into the WMMA
// A-matrix (16x32, M=query, K=class) per-lane layout, contiguous K (not ISA table order):
//   lanes 0-15 (M=lane): halves 0-15 -> K=0..15
//   lanes 16-31:         halves 0-15 -> K=16..31
// probA element index: (((qt*3 + s)*32) + lane)*16 + h
// ---------------------------------------------------------------------------
__global__ __launch_bounds__(256) void softmax_swizzle_k(
    const float* __restrict__ logits, _Float16* __restrict__ probA)
{
    __shared__ float rowbuf[8][NCP];
    const int lane = threadIdx.x & 31;
    const int wave = threadIdx.x >> 5;
    const int q    = blockIdx.x * 8 + wave;          // grid = 2400 -> q < 19200

    const float* row = logits + q * NC;
    const bool v2ok = (lane + 64) < NC;
    float v0 = row[lane];
    float v1 = row[lane + 32];
    float v2 = v2ok ? row[lane + 64] : -INFINITY;

    float m = fmaxf(fmaxf(v0, v1), v2);
    #pragma unroll
    for (int off = 16; off; off >>= 1) m = fmaxf(m, __shfl_xor(m, off, 32));

    float e0 = __expf(v0 - m);
    float e1 = __expf(v1 - m);
    float e2 = v2ok ? __expf(v2 - m) : 0.0f;
    float s  = e0 + e1 + e2;
    #pragma unroll
    for (int off = 16; off; off >>= 1) s += __shfl_xor(s, off, 32);

    const float inv = __builtin_amdgcn_rcpf(s);
    rowbuf[wave][lane]      = e0 * inv;
    rowbuf[wave][lane + 32] = e1 * inv;
    rowbuf[wave][lane + 64] = v2ok ? e2 * inv : 0.0f;   // pad classes 92..95 = 0
    __syncthreads();

    const int qt = q >> 4, mrow = q & 15;
    #pragma unroll
    for (int j = lane; j < NCP; j += 32) {
        const int sIdx = j >> 5;
        const int r    = j & 31;
        const int hi   = (r >> 4) & 1;
        const int h    = r & 15;
        const int c    = sIdx * 32 + hi * 16 + h;                  // A-layout K (contiguous)
        const int lgpr = mrow + hi * 16;
        probA[(((qt * 3 + sIdx) * 32) + lgpr) * 16 + h] = (_Float16)rowbuf[wave][c];
    }
}

// ---------------------------------------------------------------------------
// Kernel 2: one-hot(label) as the WMMA B-matrix (32x16, K=class, N=target).
// B-layout: lane n (0-15) holds column N=n, halves h -> K=h; lanes 16-31 hold
// the same columns with K = 16+h.
// ---------------------------------------------------------------------------
__global__ __launch_bounds__(96) void build_onehot_k(
    const int* __restrict__ labels, _Float16* __restrict__ onehotB)
{
    const int j  = threadIdx.x;       // 0..95
    const int s  = j >> 5;
    const int l  = j & 31;
    const int tt = blockIdx.x;        // 0..127
    const int lab  = labels[tt * 16 + (l & 15)];
    const int kofs = s * 32 + ((l >= 16) ? 16 : 0);

    v16h vals;
    #pragma unroll
    for (int h = 0; h < 16; ++h)
        vals[h] = ((kofs + h) == lab) ? (_Float16)1.0f : (_Float16)0.0f;
    *((v16h*)onehotB + (tt * 3 + s) * 32 + l) = vals;
}

// ---------------------------------------------------------------------------
// Kernel 3: main tile kernel. One wave covers 16 queries x 64 targets as
// four 16x16 tiles, reusing the A operand (f16 prob) and the query boxes
// held in registers. Class term: 3 chained v_wmma_f32_16x16x32_f16 per tile
// (K=96); L1 + GIoU on the (dual-issued) fp32 VALU with a single fused
// reciprocal per pair:
//   giou = iou - 1 + union/area_e
//   cost = 5*L1 - prob + 2 - 2*(inter*area_e + union^2) * rcp(union*area_e)
// D layout: lane l -> N(target)=l&15, VGPR v -> M(query)=v + 8*(l>=16).
// ---------------------------------------------------------------------------
__global__ __launch_bounds__(256) void matcher_main_k(
    const float*    __restrict__ pred_boxes,   // [BQ,4] cxcywh
    const float*    __restrict__ tgt_boxes,    // [NT,4] cxcywh
    const _Float16* __restrict__ probA,
    const _Float16* __restrict__ onehotB,
    float*          __restrict__ out)          // [BQ, NT]
{
    const int lane   = threadIdx.x & 31;
    const int wave   = threadIdx.x >> 5;
    const int qt     = blockIdx.x;                              // 0..1199
    const int ttbase = (blockIdx.y * 8 + wave) * TILES_PER_WAVE; // 0..124

    // ---- A operand (prob, 16 queries x 96 classes), kept in registers ----
    const v16h* Ap = (const v16h*)probA + (size_t)(qt * 3) * 32 + lane;
    const v16h a0 = Ap[0], a1 = Ap[32], a2 = Ap[64];

    // ---- Query boxes for this wave's 8 M-rows, hoisted out of tile loop ----
    const int hi    = lane >> 4;
    const int qbase = qt * 16 + hi * 8;
    float4 qb[8];
    float qx0[8], qy0[8], qx1[8], qy1[8], qarea[8];
    #pragma unroll
    for (int v = 0; v < 8; ++v) {
        qb[v] = ((const float4*)pred_boxes)[qbase + v];
        qx0[v] = qb[v].x - 0.5f * qb[v].z;  qy0[v] = qb[v].y - 0.5f * qb[v].w;
        qx1[v] = qb[v].x + 0.5f * qb[v].z;  qy1[v] = qb[v].y + 0.5f * qb[v].w;
        qarea[v] = qb[v].z * qb[v].w;
    }

    const v16h* Bp = (const v16h*)onehotB + (size_t)(ttbase * 3) * 32 + lane;

    #pragma unroll 1   // keep rolled: bounds VGPRs, preserves occupancy
    for (int i = 0; i < TILES_PER_WAVE; ++i) {
        const int tt = ttbase + i;

        const v16h b0 = Bp[0], b1 = Bp[32], b2 = Bp[64];
        Bp += 96;

        v8f acc = {};
        acc = __builtin_amdgcn_wmma_f32_16x16x32_f16(false, a0, false, b0, (short)0, acc, false, false);
        acc = __builtin_amdgcn_wmma_f32_16x16x32_f16(false, a1, false, b1, (short)0, acc, false, false);
        acc = __builtin_amdgcn_wmma_f32_16x16x32_f16(false, a2, false, b2, (short)0, acc, false, false);

        // Fixed target per lane for this tile
        const int t = tt * 16 + (lane & 15);
        const float4 tb = ((const float4*)tgt_boxes)[t];
        const float tx0 = tb.x - 0.5f * tb.z, ty0 = tb.y - 0.5f * tb.w;
        const float tx1 = tb.x + 0.5f * tb.z, ty1 = tb.y + 0.5f * tb.w;
        const float tarea = tb.z * tb.w;

        #pragma unroll
        for (int v = 0; v < 8; ++v) {
            // L1 cdist in raw cxcywh space
            const float l1 = fabsf(qb[v].x - tb.x) + fabsf(qb[v].y - tb.y)
                           + fabsf(qb[v].z - tb.z) + fabsf(qb[v].w - tb.w);

            // Intersection / union
            float iw = fminf(qx1[v], tx1) - fmaxf(qx0[v], tx0); iw = fmaxf(iw, 0.0f);
            float ih = fminf(qy1[v], ty1) - fmaxf(qy0[v], ty0); ih = fmaxf(ih, 0.0f);
            const float inter = iw * ih;
            const float uni   = qarea[v] + tarea - inter;

            // Enclosing box
            float ew = fmaxf(qx1[v], tx1) - fminf(qx0[v], tx0); ew = fmaxf(ew, 0.0f);
            float eh = fmaxf(qy1[v], ty1) - fminf(qy0[v], ty0); eh = fmaxf(eh, 0.0f);
            const float areae = ew * eh;

            // Single fused reciprocal:
            // cost = 5*L1 - prob + 2 - 2*(inter*areae + uni*uni) * rcp(uni*areae)
            const float r   = __builtin_amdgcn_rcpf(uni * areae);
            const float num = fmaf(inter, areae, uni * uni);
            const float bias = 2.0f - acc[v];
            const float cost = fmaf(-2.0f * r, num, fmaf(5.0f, l1, bias));

            out[(size_t)(qbase + v) * NT + t] = cost;
        }
    }
}

// ---------------------------------------------------------------------------
extern "C" void kernel_launch(void* const* d_in, const int* in_sizes, int n_in,
                              void* d_out, int out_size, void* d_ws, size_t ws_size,
                              hipStream_t stream) {
    const float* class_logits = (const float*)d_in[0];   // [B,Q,C]
    const float* pred_boxes   = (const float*)d_in[1];   // [B,Q,4]
    const int*   tgt_labels   = (const int*)  d_in[2];   // [T]
    const float* tgt_boxes    = (const float*)d_in[3];   // [T,4]
    float*       out          = (float*)d_out;           // [B,Q,T]

    // Workspace carving (32B aligned for v16h loads)
    _Float16* probA   = (_Float16*)d_ws;                            // 1200*3*512*2 = 3,686,400 B
    _Float16* onehotB = (_Float16*)((char*)d_ws + 3686400);         // 128*3*512*2  =   393,216 B

    softmax_swizzle_k<<<dim3(BQ / 8), dim3(256), 0, stream>>>(class_logits, probA);
    build_onehot_k  <<<dim3(TTN),    dim3(96),  0, stream>>>(tgt_labels, onehotB);

    // 128 target tiles = 4 y-blocks * 8 waves * 4 tiles/wave
    matcher_main_k  <<<dim3(QT, TTN / (8 * TILES_PER_WAVE)), 256, 0, stream>>>(
        pred_boxes, tgt_boxes, probA, onehotB, out);
}
